// BiMamba_1211180777595
// MI455X (gfx1250) — matrix operand order repaired and checked by reading, not compiled, not run on hardware
//
#include <hip/hip_runtime.h>
#include <math.h>

typedef __attribute__((ext_vector_type(16))) _Float16 v16h;
typedef __attribute__((ext_vector_type(8)))  _Float16 v8h;
typedef __attribute__((ext_vector_type(8)))  float    v8f;

#define LSEQ   2048
#define BSZ    2
#define MROWS  (BSZ * LSEQ)   // 4096
#define DMODEL 512
#define DIN    1024
#define NXZ    (2 * DIN)      // 2048
#define DST    16
#define DTR    32
#define NXD    (DTR + 2*DST)  // 64
#define LH     40             // padded LDS row stride in halves (80B, 16B-aligned rows)
#define NCH    32             // scan chunks
#define CH     (LSEQ / NCH)   // 64 steps per chunk
#define SCN    (BSZ * DIN * DST)  // per-chunk carry-state slab (32768)

__device__ inline v16h frag_cat(v8h a, v8h b) {
  return __builtin_shufflevector(a, b, 0,1,2,3,4,5,6,7,8,9,10,11,12,13,14,15);
}

// CDNA5 async global->LDS copy (VGPR-bypassing), tracked by ASYNCcnt.
__device__ inline void async_ld_b128(_Float16* lds_dst, const _Float16* gsrc) {
  uint32_t loff = (uint32_t)(uintptr_t)lds_dst;
  asm volatile("global_load_async_to_lds_b128 %0, %1, off"
               :: "v"(loff), "v"(gsrc) : "memory");
}

__device__ inline float softplusf(float x) {
  return (x > 20.f) ? x : log1pf(expf(x));
}

// C[M x N] = alpha * opA(A)[M x K] * W[N x K]^T + beta * C ; f16 WMMA, f32 accumulate.
// 128x64 tile per 128-thread block; 32x64 per wave (2 A-frags x 4 B-frags = 8 WMMA/step).
// Async double-buffered LDS staging; flipA/flipC reverse time within LSEQ segments.
__global__ __launch_bounds__(128) void k_gemm_f16(
    const _Float16* __restrict__ A, const _Float16* __restrict__ W,
    float* __restrict__ C, _Float16* __restrict__ Ch,
    int K, int N, int lda, int ldc,
    float alpha, float beta, int flipA, int flipC)
{
  __shared__ __align__(16) _Float16 As[2][128 * LH];
  __shared__ __align__(16) _Float16 Bs[2][64 * LH];
  const int tid = threadIdx.x;
  const int wv  = tid >> 5;        // wave 0..3 -> 32-row strip
  const int ln  = tid & 31;
  const int cc  = ln & 15;
  const int hi  = ln >> 4;
  const int m0  = blockIdx.y * 128;
  const int n0  = blockIdx.x * 64;

  // staging: thread t owns A row t (4x b128) and B half-row (2x b128)
  int grA = m0 + tid;
  if (flipA) { int bb = grA / LSEQ; int tt = grA - bb * LSEQ; grA = bb * LSEQ + (LSEQ - 1 - tt); }
  const _Float16* agp = A + (size_t)grA * lda;
  const _Float16* bgp = W + (size_t)(n0 + (tid >> 1)) * K + (tid & 1) * 16;
  _Float16* adst[2] = { &As[0][tid * LH], &As[1][tid * LH] };
  _Float16* bdst[2] = { &Bs[0][(tid >> 1) * LH + (tid & 1) * 16],
                        &Bs[1][(tid >> 1) * LH + (tid & 1) * 16] };

  auto issue = [&](int kt, int buf) {
    const _Float16* ag = agp + kt * 32;
    const _Float16* bg = bgp + kt * 32;
    _Float16* ad = adst[buf];
    _Float16* bd = bdst[buf];
    async_ld_b128(ad,      ag);
    async_ld_b128(ad + 8,  ag + 8);
    async_ld_b128(ad + 16, ag + 16);
    async_ld_b128(ad + 24, ag + 24);
    async_ld_b128(bd,      bg);
    async_ld_b128(bd + 8,  bg + 8);
  };

  v8f zero = {0.f,0.f,0.f,0.f,0.f,0.f,0.f,0.f};
  v8f acc0[4] = {zero, zero, zero, zero};
  v8f acc1[4] = {zero, zero, zero, zero};

  const int nK = K >> 5;
  issue(0, 0);
  #pragma unroll 2
  for (int kt = 0; kt < nK; ++kt) {
    const int cur = kt & 1;
    if (kt + 1 < nK) {
      issue(kt + 1, cur ^ 1);                           // prefetch next tile
      asm volatile("s_wait_asynccnt 0x6" ::: "memory"); // 6 newest stay in flight
    } else {
      asm volatile("s_wait_asynccnt 0x0" ::: "memory");
    }
    __syncthreads();
    // A fragment: lane (cc,hi) holds row M=cc, K = {8hi..8hi+7} U {16+8hi..}
    const v8h* ar0 = (const v8h*)&As[cur][(wv * 32 + cc) * LH];
    const v8h* ar1 = (const v8h*)&As[cur][(wv * 32 + 16 + cc) * LH];
    v16h af0 = frag_cat(ar0[hi], ar0[hi + 2]);
    v16h af1 = frag_cat(ar1[hi], ar1[hi + 2]);
    v16h bf[4];
    #pragma unroll
    for (int nt = 0; nt < 4; ++nt) {
      // B fragment: lane (cc,hi) holds col N=cc, K = {8hi..8hi+7} U {16+8hi..}
      const v8h* br = (const v8h*)&Bs[cur][(nt * 16 + cc) * LH];
      bf[nt] = frag_cat(br[hi], br[hi + 2]);
    }
    #pragma unroll
    for (int nt = 0; nt < 4; ++nt)
      acc0[nt] = __builtin_amdgcn_wmma_f32_16x16x32_f16(
          false, af0, false, bf[nt], (short)0, acc0[nt], false, false);
    #pragma unroll
    for (int nt = 0; nt < 4; ++nt)
      acc1[nt] = __builtin_amdgcn_wmma_f32_16x16x32_f16(
          false, af1, false, bf[nt], (short)0, acc1[nt], false, false);
    __syncthreads();
  }

  // C/D layout: element v of acc <-> row (v + 8*hi), col cc
  #pragma unroll
  for (int g = 0; g < 2; ++g) {
    v8f* accg = g ? acc1 : acc0;
    #pragma unroll
    for (int nt = 0; nt < 4; ++nt) {
      int col = n0 + nt * 16 + cc;
      #pragma unroll
      for (int v = 0; v < 8; ++v) {
        int m = m0 + wv * 32 + g * 16 + 8 * hi + v;
        int gr = m;
        if (flipC) { int bb = m / LSEQ; int tt = m - bb * LSEQ; gr = bb * LSEQ + (LSEQ - 1 - tt); }
        float val = alpha * accg[nt][v];
        float* pp = C + (size_t)gr * ldc + col;
        if (beta != 0.f) val += beta * (*pp);
        *pp = val;
        if (Ch) Ch[(size_t)gr * ldc + col] = (_Float16)val;
      }
    }
  }
}

// one-time f32 -> f16 conversion
__global__ __launch_bounds__(256) void k_cvt(const float* __restrict__ s,
                                             _Float16* __restrict__ d, int n) {
  int i = blockIdx.x * 256 + threadIdx.x;
  if (i < n) d[i] = (_Float16)s[i];
}

// xc = silu(conv_b + depthwise_causal_conv4(xi)); emits f32 (scan) + f16 (GEMM A)
__global__ __launch_bounds__(256) void k_conv_silu(
    const float* __restrict__ xz, const float* __restrict__ conv_w,
    const float* __restrict__ conv_b, float* __restrict__ xc,
    _Float16* __restrict__ xch)
{
  int idx = blockIdx.x * blockDim.x + threadIdx.x;   // over MROWS*DIN
  int d = idx & (DIN - 1);
  int m = idx >> 10;
  int b = m / LSEQ;
  int t = m - b * LSEQ;
  float a = conv_b[d];
  #pragma unroll
  for (int k = 0; k < 4; ++k) {
    int tt = t - 3 + k;
    float xv = (tt >= 0) ? xz[(size_t)(b * LSEQ + tt) * NXZ + d] : 0.f;
    a += xv * conv_w[d * 4 + k];
  }
  float s = a / (1.f + expf(-a));
  xc[idx] = s;
  xch[idx] = (_Float16)s;
}

// --- chunked selective scan: h_t = dA_t*h_{t-1} + dBu_t is linear, so
// h_chunk_end = (prod dA)*h_in + h_local. 3 passes cut serial depth 2048 -> ~160.

// pass 1: per-chunk local scan with h0=0; store prod(dA) and local end state.
__global__ __launch_bounds__(256) void k_scan_part(
    const float* __restrict__ dtraw, const float* __restrict__ dt_b,
    const float* __restrict__ xc,    const float* __restrict__ xdbl,
    const float* __restrict__ A_log, float* __restrict__ P, float* __restrict__ HL)
{
  const int tid = threadIdx.x;
  const int s = tid & 15, dloc = tid >> 4;
  const int d = blockIdx.x * 16 + dloc;
  const int b = blockIdx.y;
  const int ch = blockIdx.z;
  const float Ads  = -expf(A_log[d * DST + s]);
  const float bias = dt_b[d];
  float h = 0.f, prod = 1.f;
  const int t0 = ch * CH;
  for (int t = t0; t < t0 + CH; ++t) {
    size_t m = (size_t)b * LSEQ + t;
    if ((t & 7) == 0 && t + 8 < t0 + CH)
      __builtin_prefetch(&dtraw[(m + 8) * DIN + d], 0, 0);
    float dtv = softplusf(dtraw[m * DIN + d] + bias);
    float dA = expf(dtv * Ads);
    h = dA * h + dtv * xc[m * DIN + d] * xdbl[m * NXD + DTR + s];
    prod *= dA;
  }
  size_t g = (size_t)ch * SCN + ((size_t)b * DIN + d) * DST + s;
  P[g] = prod; HL[g] = h;
}

// pass 2: sequential carry combine across the 32 chunks (coalesced, trivial).
__global__ __launch_bounds__(256) void k_scan_carry(
    const float* __restrict__ P, const float* __restrict__ HL,
    float* __restrict__ Carry)
{
  int i = blockIdx.x * 256 + threadIdx.x;   // 0..SCN-1 = (b,d,s)
  float h = 0.f;
  for (int ch = 0; ch < NCH; ++ch) {
    size_t g = (size_t)ch * SCN + i;
    Carry[g] = h;                            // state entering chunk ch
    h = P[g] * h + HL[g];
  }
}

// pass 3: replay each chunk from its true incoming state; emit gated f16 output.
__global__ __launch_bounds__(256) void k_scan_final(
    const float* __restrict__ dtraw, const float* __restrict__ dt_b,
    const float* __restrict__ xc,    const float* __restrict__ xdbl,
    const float* __restrict__ A_log, const float* __restrict__ Dp,
    const float* __restrict__ xz,    const float* __restrict__ Carry,
    _Float16* __restrict__ y)
{
  const int tid = threadIdx.x;
  const int s = tid & 15, dloc = tid >> 4;
  const int d = blockIdx.x * 16 + dloc;
  const int b = blockIdx.y;
  const int ch = blockIdx.z;
  const float Ads  = -expf(A_log[d * DST + s]);
  const float bias = dt_b[d];
  const float Dd   = Dp[d];
  float h = Carry[(size_t)ch * SCN + ((size_t)b * DIN + d) * DST + s];
  const int t0 = ch * CH;
  for (int t = t0; t < t0 + CH; ++t) {
    size_t m = (size_t)b * LSEQ + t;
    if ((t & 7) == 0 && t + 8 < t0 + CH) {
      __builtin_prefetch(&dtraw[(m + 8) * DIN + d], 0, 0);
      __builtin_prefetch(&xdbl[(m + 8) * NXD + DTR + s], 0, 0);
    }
    float dtv = softplusf(dtraw[m * DIN + d] + bias);
    float xcv = xc[m * DIN + d];
    float dA = expf(dtv * Ads);
    h = dA * h + dtv * xcv * xdbl[m * NXD + DTR + s];
    float part = h * xdbl[m * NXD + DTR + DST + s];
    part += __shfl_xor(part, 1, 16);
    part += __shfl_xor(part, 2, 16);
    part += __shfl_xor(part, 4, 16);
    part += __shfl_xor(part, 8, 16);
    if (s == 0) {
      float zv = xz[m * NXZ + DIN + d];
      y[m * DIN + d] = (_Float16)((part + Dd * xcv) * (zv / (1.f + expf(-zv))));
    }
  }
}

extern "C" void kernel_launch(void* const* d_in, const int* in_sizes, int n_in,
                              void* d_out, int out_size, void* d_ws, size_t ws_size,
                              hipStream_t stream) {
  (void)in_sizes; (void)n_in; (void)out_size; (void)ws_size;
  const float* x = (const float*)d_in[0];

  float* xz    = (float*)d_ws;                        // 4096 x 2048 f32
  float* xc    = xz    + (size_t)MROWS * NXZ;         // 4096 x 1024 f32
  float* xdbl  = xc    + (size_t)MROWS * DIN;         // 4096 x 64   f32
  float* dtraw = xdbl  + (size_t)MROWS * NXD;         // 4096 x 1024 f32
  float* P     = dtraw + (size_t)MROWS * DIN;         // scan chunk products
  float* HL    = P     + (size_t)NCH * SCN;           // scan chunk local states
  float* Carry = HL    + (size_t)NCH * SCN;           // scan chunk carries
  _Float16* x_h    = (_Float16*)(Carry + (size_t)NCH * SCN);
  _Float16* xc_h   = x_h    + (size_t)MROWS * DMODEL;
  _Float16* xdbl_h = xc_h   + (size_t)MROWS * DIN;
  _Float16* y_h    = xdbl_h + (size_t)MROWS * NXD;
  _Float16* p      = y_h    + (size_t)MROWS * DIN;
  _Float16 *inw_h[2], *xpj_h[2], *dtw_h[2], *otw_h[2];
  for (int d = 0; d < 2; ++d) {
    inw_h[d] = p; p += (size_t)NXZ * DMODEL;
    xpj_h[d] = p; p += (size_t)NXD * DIN;
    dtw_h[d] = p; p += (size_t)DIN * DTR;
    otw_h[d] = p; p += (size_t)DMODEL * DIN;
  }

  // one-time f32 -> f16 staging of x and all weights
  k_cvt<<<(MROWS * DMODEL) / 256, 256, 0, stream>>>(x, x_h, MROWS * DMODEL);
  for (int d = 0; d < 2; ++d) {
    k_cvt<<<(NXZ * DMODEL) / 256, 256, 0, stream>>>((const float*)d_in[1 + d * 9], inw_h[d], NXZ * DMODEL);
    k_cvt<<<(NXD * DIN)    / 256, 256, 0, stream>>>((const float*)d_in[4 + d * 9], xpj_h[d], NXD * DIN);
    k_cvt<<<(DIN * DTR)    / 256, 256, 0, stream>>>((const float*)d_in[5 + d * 9], dtw_h[d], DIN * DTR);
    k_cvt<<<(DMODEL * DIN) / 256, 256, 0, stream>>>((const float*)d_in[9 + d * 9], otw_h[d], DMODEL * DIN);
  }

  for (int dir = 0; dir < 2; ++dir) {
    const float* conv_w = (const float*)d_in[2 + dir * 9];
    const float* conv_b = (const float*)d_in[3 + dir * 9];
    const float* dt_b   = (const float*)d_in[6 + dir * 9];
    const float* A_log  = (const float*)d_in[7 + dir * 9];
    const float* Dp     = (const float*)d_in[8 + dir * 9];

    // in_proj: xz = flip_dir(x) @ in_w^T
    k_gemm_f16<<<dim3(NXZ / 64, MROWS / 128), 128, 0, stream>>>(
        x_h, inw_h[dir], xz, nullptr, DMODEL, NXZ, DMODEL, NXZ, 1.f, 0.f, dir, 0);
    // depthwise conv + silu
    k_conv_silu<<<(MROWS * DIN) / 256, 256, 0, stream>>>(xz, conv_w, conv_b, xc, xc_h);
    // x_proj: xdbl = xc @ xproj^T (dt | B | C); dual f32+f16 store
    k_gemm_f16<<<dim3(NXD / 64, MROWS / 128), 128, 0, stream>>>(
        xc_h, xpj_h[dir], xdbl, xdbl_h, DIN, NXD, DIN, NXD, 1.f, 0.f, 0, 0);
    // dt_proj (pre-bias): dtraw = xdbl[:, :32] @ dt_w^T
    k_gemm_f16<<<dim3(DIN / 64, MROWS / 128), 128, 0, stream>>>(
        xdbl_h, dtw_h[dir], dtraw, nullptr, DTR, DIN, NXD, DIN, 1.f, 0.f, 0, 0);
    // chunked selective scan (3 passes) -> gated f16 output
    k_scan_part<<<dim3(DIN / 16, BSZ, NCH), 256, 0, stream>>>(
        dtraw, dt_b, xc, xdbl, A_log, P, HL);
    k_scan_carry<<<SCN / 256, 256, 0, stream>>>(P, HL, Carry);
    k_scan_final<<<dim3(DIN / 16, BSZ, NCH), 256, 0, stream>>>(
        dtraw, dt_b, xc, xdbl, A_log, Dp, xz, Carry, y_h);
    // out_proj: out = 0.5 * flip_dir(y @ out_w^T) (+ other direction)
    k_gemm_f16<<<dim3(DMODEL / 64, MROWS / 128), 128, 0, stream>>>(
        y_h, otw_h[dir], (float*)d_out, nullptr, DIN, DMODEL, DIN, DMODEL,
        0.5f, dir ? 1.f : 0.f, 0, dir);
  }
}
